// GraphTransformerRegression_21165598834727
// MI455X (gfx1250) — matrix operand order repaired and checked by reading, not compiled, not run on hardware
//
#include <hip/hip_runtime.h>
#include <hip/hip_bf16.h>
#include <stdint.h>

// ---------------------------------------------------------------------------
// GraphTransformerRegression on MI455X (gfx1250, wave32).
// Dense GEMMs run on v_wmma_f32_16x16x32_bf16 (bf16 in, f32 accumulate),
// one wave = 16x64 output tile (A fragment reused across 4 WMMAs per K-step).
// Edge attention uses L2-resident gathers + float atomics (q/k/v = 96MB < 192MB L2).
// ---------------------------------------------------------------------------

#define NN      32000
#define NE      512000
#define INCH    128
#define HIDC    256
#define NHEADS  8
#define DHEAD   32
#define NLAYERS 4
#define KEIG    16
#define NGRAPHS 64

typedef unsigned short u16;
typedef __attribute__((ext_vector_type(16))) __bf16 v16bf;
typedef __attribute__((ext_vector_type(8)))  float  v8f;

struct Chunk32 { uint4 a, b; };   // 32 bytes = one v16bf fragment

__device__ __forceinline__ u16 f2bf(float f) {
  unsigned u = __float_as_uint(f);
  unsigned r = (u + 0x7FFFu + ((u >> 16) & 1u)) >> 16;   // RNE
  return (u16)r;
}

__device__ __forceinline__ float atomicMaxFloat(float* addr, float val) {
  // int-trick float max: valid across mixed signs; init pattern 0xFFFFFFFF acts as -inf.
  if (val >= 0.0f) {
    int old = atomicMax((int*)addr, __float_as_int(val));
    return __int_as_float(old);
  } else {
    unsigned old = atomicMin((unsigned*)addr, (unsigned)__float_as_int(val));
    return __int_as_float((int)old);
  }
}

// ---------------------------------------------------------------------------
// Elementwise casts / padding
// ---------------------------------------------------------------------------
__global__ __launch_bounds__(256) void cast_bf16_kernel(const float* __restrict__ s,
                                                        u16* __restrict__ d, long n) {
  long i = (long)blockIdx.x * 256 + threadIdx.x;
  long stride = (long)gridDim.x * 256;
  for (; i < n; i += stride) d[i] = f2bf(s[i]);
}

__global__ __launch_bounds__(256) void pad_bf16_kernel(const float* __restrict__ s,
                                                       u16* __restrict__ d,
                                                       long rows, int incols, int outcols) {
  long n = rows * outcols;
  long i = (long)blockIdx.x * 256 + threadIdx.x;
  long stride = (long)gridDim.x * 256;
  for (; i < n; i += stride) {
    long r = i / outcols; int c = (int)(i % outcols);
    d[i] = (c < incols) ? f2bf(s[r * incols + c]) : (u16)0;
  }
}

// ---------------------------------------------------------------------------
// WMMA GEMM:  Y[N,M] = act( Xbf16[N,K] @ Wbf16[M,K]^T + bias (+ add) )
// One wave -> one 16x64 output tile (4 WMMAs per K-step sharing one A fragment).
// K multiple of 32; N multiple of 16; M multiple of 64.
// ---------------------------------------------------------------------------
__device__ __forceinline__ v16bf load_frag(const u16* p, int stride_elems) {
  Chunk32 c;
  c.a = *(const uint4*)p;
  c.b = *(const uint4*)(p + stride_elems);
  return __builtin_bit_cast(v16bf, c);
}

__global__ __launch_bounds__(256) void gemm_wmma_kernel(
    const u16* __restrict__ X, const u16* __restrict__ W,
    const float* __restrict__ bias, const float* __restrict__ add,
    float* __restrict__ Yf, u16* __restrict__ Yb,
    int N, int K, int M, int relu) {
  int wave = threadIdx.x >> 5;
  int lane = threadIdx.x & 31;
  int tile = blockIdx.x * 8 + wave;
  int mg = M >> 6;                       // 64-wide column groups
  int rowT = tile / mg, cg = tile % mg;
  if (rowT >= (N >> 4)) return;
  int row0 = rowT << 4, col0 = cg << 6;
  int r = lane & 15, hi = lane >> 4;

  v8f acc0 = {}, acc1 = {}, acc2 = {}, acc3 = {};
  for (int kk = 0; kk < K; kk += 32) {
    // A fragment: row = row0 + r; K chunks {kk+hi*8 .. +7} and {kk+16+hi*8 .. +7}
    v16bf a = load_frag(X + (size_t)(row0 + r) * K + kk + hi * 8, 16);
    // B fragments: W row (output channel) = col0 + 16*c + r; K chunks {kk+hi*8 .. +7} and {kk+16+hi*8 .. +7}
    const u16* wb = W + (size_t)(col0 + r) * K + kk + hi * 8;
    size_t wstep = (size_t)16 * K;
    v16bf b0 = load_frag(wb,             16);
    v16bf b1 = load_frag(wb + wstep,     16);
    v16bf b2 = load_frag(wb + 2 * wstep, 16);
    v16bf b3 = load_frag(wb + 3 * wstep, 16);
    acc0 = __builtin_amdgcn_wmma_f32_16x16x32_bf16(false, a, false, b0, (short)0, acc0, false, false);
    acc1 = __builtin_amdgcn_wmma_f32_16x16x32_bf16(false, a, false, b1, (short)0, acc1, false, false);
    acc2 = __builtin_amdgcn_wmma_f32_16x16x32_bf16(false, a, false, b2, (short)0, acc2, false, false);
    acc3 = __builtin_amdgcn_wmma_f32_16x16x32_bf16(false, a, false, b3, (short)0, acc3, false, false);
  }

  // D layout per 16x16 tile: VGPR j -> M = j + 8*hi ; N = lane & 15
  v8f accs[4] = {acc0, acc1, acc2, acc3};
  #pragma unroll
  for (int c = 0; c < 4; ++c) {
    int col = col0 + c * 16 + r;
    float bv = bias ? bias[col] : 0.0f;
    #pragma unroll
    for (int j = 0; j < 8; ++j) {
      int m = hi * 8 + j;
      size_t idx = (size_t)(row0 + m) * M + col;
      float vv = accs[c][j] + bv;
      if (add)  vv += add[idx];
      if (relu) vv = fmaxf(vv, 0.0f);
      if (Yf) Yf[idx] = vv;
      if (Yb) Yb[idx] = f2bf(vv);
    }
  }
}

// ---------------------------------------------------------------------------
// LayerNorm: one wave per node; output bf16 (GEMM input)
// ---------------------------------------------------------------------------
__global__ __launch_bounds__(256) void ln_kernel(const float* __restrict__ h,
                                                 const float* __restrict__ g,
                                                 const float* __restrict__ b,
                                                 u16* __restrict__ xnb, int n) {
  int node = (int)((blockIdx.x * 256 + threadIdx.x) >> 5);
  int lane = threadIdx.x & 31;
  if (node >= n) return;
  const float* row = h + (size_t)node * HIDC;
  float4 v0 = ((const float4*)row)[lane * 2];
  float4 v1 = ((const float4*)row)[lane * 2 + 1];
  float xr[8] = {v0.x, v0.y, v0.z, v0.w, v1.x, v1.y, v1.z, v1.w};
  float s = 0.f, s2 = 0.f;
  #pragma unroll
  for (int i = 0; i < 8; ++i) { s += xr[i]; s2 += xr[i] * xr[i]; }
  #pragma unroll
  for (int m = 16; m >= 1; m >>= 1) {
    s  += __shfl_xor(s,  m, 32);
    s2 += __shfl_xor(s2, m, 32);
  }
  float mu  = s * (1.0f / HIDC);
  float var = s2 * (1.0f / HIDC) - mu * mu;
  float inv = rsqrtf(var + 1e-5f);
  int c0 = lane * 8;
  unsigned w[4];
  #pragma unroll
  for (int p = 0; p < 4; ++p) {
    float a0 = (xr[2 * p]     - mu) * inv * g[c0 + 2 * p]     + b[c0 + 2 * p];
    float a1 = (xr[2 * p + 1] - mu) * inv * g[c0 + 2 * p + 1] + b[c0 + 2 * p + 1];
    w[p] = (unsigned)f2bf(a0) | ((unsigned)f2bf(a1) << 16);
  }
  uint4 q; q.x = w[0]; q.y = w[1]; q.z = w[2]; q.w = w[3];
  *(uint4*)(xnb + (size_t)node * HIDC + c0) = q;
}

// ---------------------------------------------------------------------------
// Edge attention (3 passes). One wave per edge; lane = head*4 + part.
// ---------------------------------------------------------------------------
__global__ __launch_bounds__(256) void edge_logits_kernel(
    const float* __restrict__ q, const float* __restrict__ k,
    const int* __restrict__ ei, float* __restrict__ logits,
    float* __restrict__ mbuf) {
  int e = (int)((blockIdx.x * 256 + threadIdx.x) >> 5);
  int lane = threadIdx.x & 31;
  if (e >= NE) return;
  int src = ei[e], dst = ei[NE + e];
  int head = lane >> 2, part = lane & 3;
  int off = head * DHEAD + part * 8;
  const float4* qp = (const float4*)(q + (size_t)dst * HIDC + off);
  const float4* kp = (const float4*)(k + (size_t)src * HIDC + off);
  float4 q0 = qp[0], q1 = qp[1], k0 = kp[0], k1 = kp[1];
  float d = q0.x * k0.x + q0.y * k0.y + q0.z * k0.z + q0.w * k0.w
          + q1.x * k1.x + q1.y * k1.y + q1.z * k1.z + q1.w * k1.w;
  d += __shfl_xor(d, 1, 32);
  d += __shfl_xor(d, 2, 32);
  if (part == 0) {
    float lg = d * 0.17677669529663687f;   // 1/sqrt(32)
    logits[(size_t)e * NHEADS + head] = lg;
    atomicMaxFloat(mbuf + (size_t)dst * NHEADS + head, lg);
  }
}

__global__ __launch_bounds__(256) void edge_expsum_kernel(
    const float* __restrict__ logits, const float* __restrict__ mbuf,
    const int* __restrict__ ei, float* __restrict__ sbuf) {
  long idx = (long)blockIdx.x * 256 + threadIdx.x;
  if (idx >= (long)NE * NHEADS) return;
  int e = (int)(idx >> 3), hd = (int)(idx & 7);
  int dst = ei[NE + e];
  float p = expf(logits[idx] - mbuf[(size_t)dst * NHEADS + hd]);
  atomicAdd(sbuf + (size_t)dst * NHEADS + hd, p);
}

__global__ __launch_bounds__(256) void edge_message_kernel(
    const float* __restrict__ v, const float* __restrict__ logits,
    const float* __restrict__ mbuf, const float* __restrict__ sbuf,
    const int* __restrict__ ei, float* __restrict__ outb) {
  int e = (int)((blockIdx.x * 256 + threadIdx.x) >> 5);
  int lane = threadIdx.x & 31;
  if (e >= NE) return;
  int src = ei[e], dst = ei[NE + e];
  int head = lane >> 2, part = lane & 3;
  int off = head * DHEAD + part * 8;
  float lg = logits[(size_t)e * NHEADS + head];
  float mm = mbuf[(size_t)dst * NHEADS + head];
  float ss = sbuf[(size_t)dst * NHEADS + head];
  float alpha = expf(lg - mm) / fmaxf(ss, 1e-16f);
  const float4* vp = (const float4*)(v + (size_t)src * HIDC + off);
  float4 v0 = vp[0], v1 = vp[1];
  float* op = outb + (size_t)dst * HIDC + off;
  atomicAdd(op + 0, alpha * v0.x); atomicAdd(op + 1, alpha * v0.y);
  atomicAdd(op + 2, alpha * v0.z); atomicAdd(op + 3, alpha * v0.w);
  atomicAdd(op + 4, alpha * v1.x); atomicAdd(op + 5, alpha * v1.y);
  atomicAdd(op + 6, alpha * v1.z); atomicAdd(op + 7, alpha * v1.w);
}

// ---------------------------------------------------------------------------
// Beta gate + ReLU + residual (in-place update of h). One wave per node.
// ---------------------------------------------------------------------------
__global__ __launch_bounds__(256) void beta_kernel(
    float* __restrict__ h, const float* __restrict__ outb,
    const float* __restrict__ skip, const float* __restrict__ wb, int n) {
  int node = (int)((blockIdx.x * 256 + threadIdx.x) >> 5);
  int lane = threadIdx.x & 31;
  if (node >= n) return;
  int c0 = lane * 8;
  size_t base = (size_t)node * HIDC + c0;
  float4 o0 = *(const float4*)(outb + base);
  float4 o1 = *(const float4*)(outb + base + 4);
  float4 s0 = *(const float4*)(skip + base);
  float4 s1 = *(const float4*)(skip + base + 4);
  float o[8] = {o0.x, o0.y, o0.z, o0.w, o1.x, o1.y, o1.z, o1.w};
  float sk[8] = {s0.x, s0.y, s0.z, s0.w, s1.x, s1.y, s1.z, s1.w};
  float acc = 0.f;
  #pragma unroll
  for (int i = 0; i < 8; ++i) {
    acc += o[i] * wb[c0 + i] + sk[i] * wb[HIDC + c0 + i]
         + (o[i] - sk[i]) * wb[2 * HIDC + c0 + i];
  }
  #pragma unroll
  for (int m = 16; m >= 1; m >>= 1) acc += __shfl_xor(acc, m, 32);
  float beta = 1.0f / (1.0f + expf(-acc));
  #pragma unroll
  for (int i = 0; i < 8; ++i) {
    float y = beta * sk[i] + (1.0f - beta) * o[i];
    h[base + i] = fmaxf(y, 0.0f) + h[base + i];
  }
}

// ---------------------------------------------------------------------------
// Graph pooling + final MLP
// ---------------------------------------------------------------------------
__global__ __launch_bounds__(256) void pool_kernel(const float* __restrict__ h,
                                                   const int* __restrict__ batch,
                                                   float* __restrict__ xg,
                                                   float* __restrict__ counts, int n) {
  int node = (int)((blockIdx.x * 256 + threadIdx.x) >> 5);
  int lane = threadIdx.x & 31;
  if (node >= n) return;
  int g = batch[node];
  int c0 = lane * 8;
  const float* row = h + (size_t)node * HIDC + c0;
  float* dstp = xg + (size_t)g * HIDC + c0;
  #pragma unroll
  for (int i = 0; i < 8; ++i) atomicAdd(dstp + i, row[i]);
  if (lane == 0) atomicAdd(counts + g, 1.0f);
}

__global__ __launch_bounds__(256) void final_mlp_kernel(
    const float* __restrict__ xg, const float* __restrict__ counts,
    const float* __restrict__ w1, const float* __restrict__ b1,
    const float* __restrict__ w2, const float* __restrict__ b2,
    float* __restrict__ out) {
  __shared__ float xr[HIDC];
  __shared__ float red[HIDC];
  int g = blockIdx.x, t = threadIdx.x;
  float cnt = fmaxf(counts[g], 1.0f);
  xr[t] = xg[(size_t)g * HIDC + t] / cnt;
  __syncthreads();
  float acc = b1[t];
  const float* wr = w1 + (size_t)t * HIDC;
  for (int i = 0; i < HIDC; ++i) acc += xr[i] * wr[i];
  red[t] = fmaxf(acc, 0.0f) * w2[t];
  __syncthreads();
  for (int s = 128; s > 0; s >>= 1) {
    if (t < s) red[t] += red[t + s];
    __syncthreads();
  }
  if (t == 0) out[g] = red[0] + b2[0];
}

// ---------------------------------------------------------------------------
// Host orchestration
// ---------------------------------------------------------------------------
static void launch_gemm(const u16* X, const u16* W, const float* bias,
                        const float* add, float* Yf, u16* Yb,
                        int N, int K, int M, int relu, hipStream_t st) {
  int tiles = (N / 16) * (M / 64);
  int blocks = (tiles + 7) / 8;
  gemm_wmma_kernel<<<blocks, 256, 0, st>>>(X, W, bias, add, Yf, Yb, N, K, M, relu);
}

static void launch_cast(const float* s, u16* d, long n, hipStream_t st) {
  long blocks = (n + 255) / 256;
  if (blocks > 8192) blocks = 8192;
  cast_bf16_kernel<<<(int)blocks, 256, 0, st>>>(s, d, n);
}

extern "C" void kernel_launch(void* const* d_in, const int* in_sizes, int n_in,
                              void* d_out, int out_size, void* d_ws, size_t ws_size,
                              hipStream_t stream) {
  const float* x      = (const float*)d_in[0];
  const float* lpe    = (const float*)d_in[1];
  const float* pe_w1  = (const float*)d_in[2];
  const float* pe_b1  = (const float*)d_in[3];
  const float* pe_w2  = (const float*)d_in[4];
  const float* pe_b2  = (const float*)d_in[5];
  const float* emb_w  = (const float*)d_in[6];
  const float* emb_b  = (const float*)d_in[7];
  const float* ln_g   = (const float*)d_in[8];
  const float* ln_b   = (const float*)d_in[9];
  const float* wq     = (const float*)d_in[10];
  const float* bq     = (const float*)d_in[11];
  const float* wk     = (const float*)d_in[12];
  const float* bk     = (const float*)d_in[13];
  const float* wv     = (const float*)d_in[14];
  const float* bv     = (const float*)d_in[15];
  const float* wskip  = (const float*)d_in[16];
  const float* bskip  = (const float*)d_in[17];
  const float* wbeta  = (const float*)d_in[18];
  const float* mlp_w1 = (const float*)d_in[19];
  const float* mlp_b1 = (const float*)d_in[20];
  const float* mlp_w2 = (const float*)d_in[21];
  const float* mlp_b2 = (const float*)d_in[22];
  const int*   ei     = (const int*)d_in[23];
  const int*   batch  = (const int*)d_in[24];
  float* out = (float*)d_out;

  // ---- workspace bump allocator ----
  size_t off = 0;
  auto alloc = [&](size_t bytes) -> void* {
    void* p = (char*)d_ws + off;
    off += (bytes + 255) & ~(size_t)255;
    return p;
  };
  float* h     = (float*)alloc((size_t)NN * HIDC * 4);
  u16*   xnb   = (u16*)  alloc((size_t)NN * HIDC * 2);
  float* qf    = (float*)alloc((size_t)NN * HIDC * 4);
  float* kf    = (float*)alloc((size_t)NN * HIDC * 4);
  float* vf    = (float*)alloc((size_t)NN * HIDC * 4);
  float* skipf = (float*)alloc((size_t)NN * HIDC * 4);
  float* outb  = (float*)alloc((size_t)NN * HIDC * 4);
  float* logit = (float*)alloc((size_t)NE * NHEADS * 4);
  float* mbuf  = (float*)alloc((size_t)NN * NHEADS * 4);
  float* sbuf  = (float*)alloc((size_t)NN * NHEADS * 4);
  float* xg    = (float*)alloc((size_t)NGRAPHS * HIDC * 4);
  float* cnts  = (float*)alloc((size_t)NGRAPHS * 4);
  u16* wqb     = (u16*)alloc((size_t)NLAYERS * HIDC * HIDC * 2);
  u16* wkb     = (u16*)alloc((size_t)NLAYERS * HIDC * HIDC * 2);
  u16* wvb     = (u16*)alloc((size_t)NLAYERS * HIDC * HIDC * 2);
  u16* wsb     = (u16*)alloc((size_t)NLAYERS * HIDC * HIDC * 2);
  u16* pw1b    = (u16*)alloc((size_t)HIDC * 32 * 2);
  u16* pw2b    = (u16*)alloc((size_t)HIDC * HIDC * 2);
  u16* embb    = (u16*)alloc((size_t)HIDC * INCH * 2);
  // Prologue-only buffers alias layer buffers (dead before layer 0 uses them)
  u16*   xb    = (u16*)qf;     // 8.2MB  <= 32.8MB
  u16*   lpeb  = (u16*)kf;     // 2.0MB  <= 32.8MB
  u16*   pehb  = xnb;          // 16.4MB == 16.4MB
  float* peout = outb;         // 32.8MB == 32.8MB

  // ---- prologue: weight conversion + embeddings ----
  launch_cast(wq,    wqb,  (long)NLAYERS * HIDC * HIDC, stream);
  launch_cast(wk,    wkb,  (long)NLAYERS * HIDC * HIDC, stream);
  launch_cast(wv,    wvb,  (long)NLAYERS * HIDC * HIDC, stream);
  launch_cast(wskip, wsb,  (long)NLAYERS * HIDC * HIDC, stream);
  launch_cast(pe_w2, pw2b, (long)HIDC * HIDC, stream);
  launch_cast(emb_w, embb, (long)HIDC * INCH, stream);
  launch_cast(x,     xb,   (long)NN * INCH, stream);
  pad_bf16_kernel<<<4000, 256, 0, stream>>>(lpe,   lpeb, NN,   KEIG, 32);
  pad_bf16_kernel<<<32,   256, 0, stream>>>(pe_w1, pw1b, HIDC, KEIG, 32);

  // pe hidden = relu(lpe_pad @ pw1_pad^T + b1)  -> bf16
  launch_gemm(lpeb, pw1b, pe_b1, nullptr, nullptr, pehb, NN, 32, HIDC, 1, stream);
  // pe = peh @ pw2^T + b2
  launch_gemm(pehb, pw2b, pe_b2, nullptr, peout, nullptr, NN, HIDC, HIDC, 0, stream);
  // h = x @ emb_w^T + emb_b + pe
  launch_gemm(xb, embb, emb_b, peout, h, nullptr, NN, INCH, HIDC, 0, stream);

  // ---- transformer layers ----
  const int nodeBlocks = NN / 8;               // wave-per-node kernels
  const int edgeBlocks = NE / 8;               // wave-per-edge kernels
  const int ehBlocks   = (NE * NHEADS) / 256;
  for (int l = 0; l < NLAYERS; ++l) {
    ln_kernel<<<nodeBlocks, 256, 0, stream>>>(h, ln_g + l * HIDC, ln_b + l * HIDC,
                                              xnb, NN);
    size_t wOff = (size_t)l * HIDC * HIDC;
    launch_gemm(xnb, wqb + wOff, bq + l * HIDC, nullptr, qf,    nullptr, NN, HIDC, HIDC, 0, stream);
    launch_gemm(xnb, wkb + wOff, bk + l * HIDC, nullptr, kf,    nullptr, NN, HIDC, HIDC, 0, stream);
    launch_gemm(xnb, wvb + wOff, bv + l * HIDC, nullptr, vf,    nullptr, NN, HIDC, HIDC, 0, stream);
    launch_gemm(xnb, wsb + wOff, bskip + l * HIDC, nullptr, skipf, nullptr, NN, HIDC, HIDC, 0, stream);

    hipMemsetAsync(mbuf, 0xFF, (size_t)NN * NHEADS * 4, stream);  // -NaN bits == -inf under int-trick
    hipMemsetAsync(sbuf, 0x00, (size_t)NN * NHEADS * 4, stream);
    hipMemsetAsync(outb, 0x00, (size_t)NN * HIDC * 4, stream);

    edge_logits_kernel<<<edgeBlocks, 256, 0, stream>>>(qf, kf, ei, logit, mbuf);
    edge_expsum_kernel<<<ehBlocks, 256, 0, stream>>>(logit, mbuf, ei, sbuf);
    edge_message_kernel<<<edgeBlocks, 256, 0, stream>>>(vf, logit, mbuf, sbuf, ei, outb);

    beta_kernel<<<nodeBlocks, 256, 0, stream>>>(h, outb, skipf,
                                                wbeta + (size_t)l * 3 * HIDC, NN);
  }

  // ---- pooling + final MLP ----
  hipMemsetAsync(xg,   0, (size_t)NGRAPHS * HIDC * 4, stream);
  hipMemsetAsync(cnts, 0, (size_t)NGRAPHS * 4, stream);
  pool_kernel<<<nodeBlocks, 256, 0, stream>>>(h, batch, xg, cnts, NN);
  final_mlp_kernel<<<NGRAPHS, 256, 0, stream>>>(xg, cnts, mlp_w1, mlp_b1,
                                                mlp_w2, mlp_b2, out);
}
